// TokenEncoder_66692252172876
// MI455X (gfx1250) — compile-verified, matrix operand order checked by reading, not run on hardware
//
#include <hip/hip_runtime.h>

// ---------------------------------------------------------------------------
// Problem constants (match the reference)
// ---------------------------------------------------------------------------
#define BB 8
#define LL 2048
#define DD 256            // input dim (K)
#define DM 1024           // d_model (N)
#define NSIG 16
#define NTOK (BB * LL)            // 16384
#define NROW (BB * (LL + 1))      // 16392 output rows
#define MAX_TILES 1024            // worst case: all tokens in one signal
#define ROLE_OUTPUT 2

#define W_ELEMS (NSIG * DM * DD)  // 4,194,304
#define E_ELEMS (NTOK * DD)       // 4,194,304

typedef __attribute__((ext_vector_type(16))) __bf16 v16bf;
typedef __attribute__((ext_vector_type(8)))  float  v8f;

// bool inputs arrive as 1-byte numpy bools
typedef unsigned char mask_t;

// ---------------------------------------------------------------------------
// fp32 -> bf16 (RNE) + hi/lo split helpers (used only in the pre-pass)
// ---------------------------------------------------------------------------
__device__ __forceinline__ unsigned short f32_bf16_rne(float f) {
    unsigned u = __float_as_uint(f);
    u += 0x7FFFu + ((u >> 16) & 1u);
    return (unsigned short)(u >> 16);
}
__device__ __forceinline__ float bf16_f32(unsigned short h) {
    return __uint_as_float(((unsigned)h) << 16);
}
__device__ __forceinline__ void split2(float f, unsigned short& hi, unsigned short& lo) {
    hi = f32_bf16_rne(f);
    lo = f32_bf16_rne(f - bf16_f32(hi));
}

// ---------------------------------------------------------------------------
// Kernel 0: zero the per-signal counters in workspace
// ---------------------------------------------------------------------------
__global__ void tenc_zero_counts(int* __restrict__ cnt) {
    if (threadIdx.x < NSIG) cnt[threadIdx.x] = 0;
}

// ---------------------------------------------------------------------------
// Kernel 1: bucket unmasked token indices by signal id (compaction).
// List order is non-deterministic but each token lands in exactly one slot and
// its output is independent of list order -> output values deterministic.
// ---------------------------------------------------------------------------
__global__ void tenc_build_lists(const int* __restrict__ sid,
                                 const mask_t* __restrict__ mask,
                                 int* __restrict__ cnt,
                                 int* __restrict__ list) {
    int t = blockIdx.x * blockDim.x + threadIdx.x;
    if (t >= NTOK) return;
    if (!mask[t]) return;
    int s = sid[t];
    int slot = atomicAdd(&cnt[s], 1);
    list[s * NTOK + slot] = t;
}

// ---------------------------------------------------------------------------
// Kernel 2: streaming fp32 -> bf16 hi/lo split (pre-pass, done once per
// launch for proj_W and emb). 8 floats per thread: 2x b128 in, 2x b128 out.
// ---------------------------------------------------------------------------
__global__ void tenc_split_bf16(const float* __restrict__ src,
                                unsigned short* __restrict__ hi,
                                unsigned short* __restrict__ lo,
                                int n8) {
    int i = blockIdx.x * blockDim.x + threadIdx.x;
    if (i >= n8) return;
    const float4* s4 = (const float4*)src;
    float4 a = s4[2 * i], b = s4[2 * i + 1];
    float f[8] = {a.x, a.y, a.z, a.w, b.x, b.y, b.z, b.w};
    unsigned short h[8], l[8];
#pragma unroll
    for (int j = 0; j < 8; ++j) split2(f[j], h[j], l[j]);
    uint4 ph, pl;
    ph.x = (unsigned)h[0] | ((unsigned)h[1] << 16);
    ph.y = (unsigned)h[2] | ((unsigned)h[3] << 16);
    ph.z = (unsigned)h[4] | ((unsigned)h[5] << 16);
    ph.w = (unsigned)h[6] | ((unsigned)h[7] << 16);
    pl.x = (unsigned)l[0] | ((unsigned)l[1] << 16);
    pl.y = (unsigned)l[2] | ((unsigned)l[3] << 16);
    pl.z = (unsigned)l[4] | ((unsigned)l[5] << 16);
    pl.w = (unsigned)l[6] | ((unsigned)l[7] << 16);
    ((uint4*)hi)[i] = ph;
    ((uint4*)lo)[i] = pl;
}

// ---------------------------------------------------------------------------
// Kernel 3: baseline rows. Writes CLS rows and masked-token rows
// (pos/id/role embeddings only; zero projection), plus attn_keep for all rows.
// Unmasked rows are written (fully) by the WMMA kernel -> each output row is
// written exactly once per launch.
// ---------------------------------------------------------------------------
__global__ void tenc_embed_base(const int* __restrict__ pos,
                                const int* __restrict__ sid,
                                const int* __restrict__ role,
                                const mask_t* __restrict__ mask,
                                const float* __restrict__ cls_content,
                                const float* __restrict__ pos_embed,
                                const float* __restrict__ id_embed,
                                const float* __restrict__ role_embed,
                                float* __restrict__ out,
                                float* __restrict__ attn) {
    int row = blockIdx.x;                 // 0 .. NROW-1
    int b   = row / (LL + 1);
    int l   = row - b * (LL + 1);
    bool isCls = (l == 0);
    int t = b * LL + (l - 1);

    int p, si, ro; bool m;
    if (isCls) { p = 0; si = NSIG; ro = ROLE_OUTPUT; m = true; }
    else       { p = pos[t]; si = sid[t]; ro = role[t]; m = (mask[t] != 0); }

    if (threadIdx.x == 0) attn[row] = m ? 1.0f : 0.0f;
    if (!isCls && m) return;              // WMMA kernel owns this row

    const float4* pe = (const float4*)(pos_embed  + (size_t)p  * DM);
    const float4* ie = (const float4*)(id_embed   + (size_t)si * DM);
    const float4* re = (const float4*)(role_embed + (size_t)ro * DM);
    const float4* cc = (const float4*)cls_content;
    float4*       o  = (float4*)(out + (size_t)row * DM);

    int c = threadIdx.x;                  // 256 threads x float4 = 1024 cols
    float4 v = pe[c], w = ie[c], x = re[c];
    float4 r;
    r.x = v.x + w.x + x.x; r.y = v.y + w.y + x.y;
    r.z = v.z + w.z + x.z; r.w = v.w + w.w + x.w;
    if (isCls) {
        float4 cv = cc[c];
        r.x += cv.x; r.y += cv.y; r.z += cv.z; r.w += cv.w;
    }
    o[c] = r;
}

// ---------------------------------------------------------------------------
// Kernel 4: grouped projection via bf16 WMMA with pre-split hi/lo operands.
// grid.x = NSIG * MAX_TILES (signal, 16-token tile), grid.y = DM/128.
// Block = 256 threads = 8 wave32; wave w computes the 16x16 tile at
// columns blockIdx.y*128 + w*16. K loop: 256 in steps of 32.
// Inner loop per K-step: 4x ds_load_b128 (A), 4x global_load_b128 (B, L2-hot),
// 3x v_wmma_f32_16x16x32_bf16. No per-step conversion VALU.
// ---------------------------------------------------------------------------
__global__ __launch_bounds__(256)
void tenc_grouped_proj_wmma(const unsigned short* __restrict__ Ehi,
                            const unsigned short* __restrict__ Elo,
                            const unsigned short* __restrict__ Whi,
                            const unsigned short* __restrict__ Wlo,
                            const int* __restrict__ pos,
                            const int* __restrict__ role,
                            const float* __restrict__ proj_b,   // (S, DM)
                            const float* __restrict__ pos_embed,
                            const float* __restrict__ id_embed,
                            const float* __restrict__ role_embed,
                            const int* __restrict__ cnt,
                            const int* __restrict__ list,
                            float* __restrict__ out) {
    const int s    = blockIdx.x >> 10;       // signal id
    const int tile = blockIdx.x & (MAX_TILES - 1);
    const int count = cnt[s];
    if (tile * 16 >= count) return;          // uniform early exit

    __shared__ int sTok[16], sPos[16], sRole[16];
    // +8 shorts of row padding (16B) to spread LDS banks across the 16 rows
    __shared__ __align__(16) unsigned short sAhi[16][DD + 8];
    __shared__ __align__(16) unsigned short sAlo[16][DD + 8];

    const int tid = threadIdx.x;
    if (tid < 16) {
        int idx = tile * 16 + tid;
        int t = (idx < count) ? list[s * NTOK + idx] : -1;
        sTok[tid]  = t;
        sPos[tid]  = (t >= 0) ? pos[t]  : 0;
        sRole[tid] = (t >= 0) ? role[t] : 0;
    }
    __syncthreads();

    // Stage A: 16 gathered pre-split bf16 rows -> LDS (pure b128 copies).
    {
        int r  = tid >> 4;
        int d0 = (tid & 15) * 16;            // 16 shorts per thread
        int t  = sTok[r];
        const uint4 z = make_uint4(0u, 0u, 0u, 0u);
        const uint4* eh = (const uint4*)(Ehi + (size_t)(t < 0 ? 0 : t) * DD);
        const uint4* el = (const uint4*)(Elo + (size_t)(t < 0 ? 0 : t) * DD);
        int q = d0 >> 3;                      // uint4 index (8 shorts each)
        uint4 h0 = (t >= 0) ? eh[q]     : z;
        uint4 h1 = (t >= 0) ? eh[q + 1] : z;
        uint4 l0 = (t >= 0) ? el[q]     : z;
        uint4 l1 = (t >= 0) ? el[q + 1] : z;
        *(uint4*)&sAhi[r][d0]     = h0;
        *(uint4*)&sAhi[r][d0 + 8] = h1;
        *(uint4*)&sAlo[r][d0]     = l0;
        *(uint4*)&sAlo[r][d0 + 8] = l1;
    }
    __syncthreads();

    const int lane = tid & 31;
    const int wv   = tid >> 5;
    const int nloc = lane & 15;                 // A row M / B column N
    const int kb   = (lane < 16) ? 0 : 8;       // K sub-chunk per half-wave
    const int col0 = blockIdx.y * 128 + wv * 16;
    const int col  = col0 + nloc;

    // B fragment source rows (pre-split bf16, contiguous in K; L2-resident).
    const unsigned short* whrow = Whi + ((size_t)s * DM + col) * DD;
    const unsigned short* wlrow = Wlo + ((size_t)s * DM + col) * DD;

    v8f acc = {};
#pragma unroll
    for (int k0 = 0; k0 < DD; k0 += 32) {
        // A fragments: two ds_load_b128 each, per ISA 16-bit A layout
        union { v16bf v; uint4 q[2]; } aH, aL;
        aH.q[0] = *(const uint4*)&sAhi[nloc][k0 + kb];
        aH.q[1] = *(const uint4*)&sAhi[nloc][k0 + kb + 16];
        aL.q[0] = *(const uint4*)&sAlo[nloc][k0 + kb];
        aL.q[1] = *(const uint4*)&sAlo[nloc][k0 + kb + 16];

        // B fragments: straight bf16 b128 loads, no conversion
        union { v16bf v; uint4 q[2]; } bH, bL;
        bH.q[0] = *(const uint4*)(whrow + k0 + kb);
        bH.q[1] = *(const uint4*)(whrow + k0 + kb + 16);
        bL.q[0] = *(const uint4*)(wlrow + k0 + kb);
        bL.q[1] = *(const uint4*)(wlrow + k0 + kb + 16);
        if (k0 + 32 < DD) {
            __builtin_prefetch(whrow + k0 + 32 + kb, 0, 1);
            __builtin_prefetch(wlrow + k0 + 32 + kb, 0, 1);
        }

        // fp32-accurate product: hi*hi + hi*lo + lo*hi (lo*lo negligible)
        acc = __builtin_amdgcn_wmma_f32_16x16x32_bf16(
                  false, aH.v, false, bH.v, (short)0, acc, false, false);
        acc = __builtin_amdgcn_wmma_f32_16x16x32_bf16(
                  false, aH.v, false, bL.v, (short)0, acc, false, false);
        acc = __builtin_amdgcn_wmma_f32_16x16x32_bf16(
                  false, aL.v, false, bH.v, (short)0, acc, false, false);
    }

    // Epilogue: C layout -> lane holds column `col`, rows mbase..mbase+7.
    const int mbase = (lane < 16) ? 0 : 8;
    const float bias = proj_b[(size_t)s * DM + col];
    const float idv  = id_embed[(size_t)s * DM + col];
#pragma unroll
    for (int v = 0; v < 8; ++v) {
        int m = mbase + v;
        int t = sTok[m];
        if (t < 0) continue;
        int row = t + (t >> 11) + 1;            // b*(L+1) + l + 1
        float addv = bias + idv
                   + pos_embed[(size_t)sPos[m]  * DM + col]
                   + role_embed[(size_t)sRole[m] * DM + col];
        out[(size_t)row * DM + col] = acc[v] + addv;
    }
}

// ---------------------------------------------------------------------------
// Launch
// ---------------------------------------------------------------------------
extern "C" void kernel_launch(void* const* d_in, const int* in_sizes, int n_in,
                              void* d_out, int out_size, void* d_ws, size_t ws_size,
                              hipStream_t stream) {
    const float*  emb         = (const float*)d_in[0];
    const int*    pos         = (const int*)d_in[1];
    const int*    sid         = (const int*)d_in[2];
    const int*    role        = (const int*)d_in[3];
    const mask_t* mask        = (const mask_t*)d_in[4];
    const float*  proj_W      = (const float*)d_in[5];
    const float*  proj_b      = (const float*)d_in[6];
    const float*  cls_content = (const float*)d_in[7];
    const float*  pos_embed   = (const float*)d_in[8];
    const float*  id_embed    = (const float*)d_in[9];
    const float*  role_embed  = (const float*)d_in[10];

    float* out  = (float*)d_out;
    float* attn = out + (size_t)NROW * DM;     // tuple output: tokens, attn_keep

    // Workspace layout (all 16B aligned):
    //   cnt   : 16 int
    //   list  : NSIG*NTOK int              (1 MB)
    //   Whi/Wlo : W_ELEMS bf16 each        (8 MB each)
    //   Ehi/Elo : E_ELEMS bf16 each        (8 MB each)
    unsigned char* wsb = (unsigned char*)d_ws;
    int* cnt  = (int*)wsb;
    int* list = cnt + NSIG;
    unsigned short* Whi = (unsigned short*)(wsb + 64 + (size_t)NSIG * NTOK * 4);
    unsigned short* Wlo = Whi + (size_t)W_ELEMS;
    unsigned short* Ehi = Wlo + (size_t)W_ELEMS;
    unsigned short* Elo = Ehi + (size_t)E_ELEMS;

    tenc_zero_counts<<<1, 32, 0, stream>>>(cnt);
    tenc_build_lists<<<NTOK / 256, 256, 0, stream>>>(sid, mask, cnt, list);
    tenc_split_bf16<<<(W_ELEMS / 8 + 255) / 256, 256, 0, stream>>>(proj_W, Whi, Wlo, W_ELEMS / 8);
    tenc_split_bf16<<<(E_ELEMS / 8 + 255) / 256, 256, 0, stream>>>(emb, Ehi, Elo, E_ELEMS / 8);
    tenc_embed_base<<<NROW, 256, 0, stream>>>(pos, sid, role, mask, cls_content,
                                              pos_embed, id_embed, role_embed,
                                              out, attn);
    tenc_grouped_proj_wmma<<<dim3(NSIG * MAX_TILES, DM / 128), 256, 0, stream>>>(
        Ehi, Elo, Whi, Wlo, pos, role, proj_b, pos_embed, id_embed, role_embed,
        cnt, list, out);
}
